// ModernBertAttention_12352325943675
// MI455X (gfx1250) — compile-verified, matrix operand order checked by reading, not run on hardware
//
#include <hip/hip_runtime.h>

#define BS 2
#define SEQ 2048
#define DIM 768
#define NH 12
#define HD 64
#define NQKV (3 * DIM)
#define WINDOW 64
#define NEGBIG (-1e9f)

typedef __bf16 bf16_t;
typedef __bf16 v16bf __attribute__((ext_vector_type(16)));
typedef float v8f __attribute__((ext_vector_type(8)));
typedef unsigned int v4u __attribute__((ext_vector_type(4)));
typedef int v8i __attribute__((ext_vector_type(8)));
typedef int v4i __attribute__((ext_vector_type(4)));

union FragBF { v16bf v; bf16_t h[16]; };
union AccF   { v8f v; float f[8]; };

#define WMMA_BF16(a, b, c) \
  __builtin_amdgcn_wmma_f32_16x16x32_bf16(false, (a), false, (b), (short)0, (c), false, false)

// A-fragment (16x32, M x K) from row-major [row][k] memory, stride in elements.
// Also serves as B-fragment when B is stored N-major ([n][k]): lane%16 -> n.
// Per ISA 7.12.2 (16-bit A 16x32): lane<16 holds K {0..7,16..23}, lane>=16 holds {8..15,24..31}.
__device__ inline v16bf frag_rowmajor(const bf16_t* base, int stride, int lane) {
  FragBF f;
  const int r = lane & 15;
  const int half = (lane >> 4) & 1;
  const bf16_t* p = base + (size_t)r * stride;
#pragma unroll
  for (int v = 0; v < 8; ++v) {
    int k = ((v >> 2) << 4) + (half << 3) + ((v & 3) << 1);
    f.h[2 * v]     = p[k];
    f.h[2 * v + 1] = p[k + 1];
  }
  return f.v;
}

// TDM: DMA a 2-D bf16 tile (tile_d0 contiguous elems x tile_d1 rows, row stride
// stride_elems) from global memory into LDS (packed, tile_d0*2 bytes per row).
// D# bitfields per CDNA5 ISA; groups 2-3 zero (2-D).
// 6-arg builtin form (clang-23 / therock headers): extra zero int32x8 group + cpol.
__device__ inline void tdm_load_tile_bf16(unsigned lds_byte_addr, const bf16_t* gsrc,
                                          unsigned tile_d0, unsigned tile_d1,
                                          unsigned stride_elems) {
  unsigned long long ga = (unsigned long long)(uintptr_t)gsrc;
  v4u g0;
  g0[0] = 1u;                                                // count=1, user D#
  g0[1] = lds_byte_addr;                                     // lds_addr
  g0[2] = (unsigned)ga;                                      // global_addr[31:0]
  g0[3] = (unsigned)((ga >> 32) & 0x01FFFFFFu) | 0x80000000u; // addr[56:32] | type=2
  v8i g1;
  g1[0] = (int)(1u << 16);                                   // data_size=1 (2 bytes)
  g1[1] = (int)((stride_elems & 0xFFFFu) << 16);             // tensor_dim0[15:0]
  g1[2] = (int)(((stride_elems >> 16) & 0xFFFFu) | (4096u << 16)); // dim0 hi | tensor_dim1=4096
  g1[3] = (int)((tile_d0 & 0xFFFFu) << 16);                  // tensor_dim1 hi=0 | tile_dim0
  g1[4] = (int)(tile_d1 & 0xFFFFu);                          // tile_dim1 | tile_dim2=0
  g1[5] = (int)stride_elems;                                 // tensor_dim0_stride[31:0]
  g1[6] = 0;                                                 // stride hi | dim1_stride lo
  g1[7] = 0;
  v4i gz4 = {0, 0, 0, 0};
  v8i gz8 = {0, 0, 0, 0, 0, 0, 0, 0};
  __builtin_amdgcn_tensor_load_to_lds(g0, g1, gz4, gz4, gz8, 0);
}

__global__ void k_cvt_f32_bf16(const float* __restrict__ src, bf16_t* __restrict__ dst, int n) {
  int i = blockIdx.x * blockDim.x + threadIdx.x;
  if (i < n) dst[i] = (bf16_t)src[i];
}

// RoPE tables: cosT/sinT[s*32 + j] = cos/sin(s * 10000^(-j/32)), j in [0,32)
__global__ void k_rope_tables(float* __restrict__ cosT, float* __restrict__ sinT) {
  int i = blockIdx.x * blockDim.x + threadIdx.x;  // SEQ*32 total
  int s = i >> 5, j = i & 31;
  float inv = __powf(10000.0f, -(float)j * (1.0f / 32.0f));
  float sn, cs;
  __sincosf((float)s * inv, &sn, &cs);
  cosT[i] = cs;
  sinT[i] = sn;
}

// Double-buffered GEMM mainloop: C(128x64) tile at (m0,n0); A [M][DIM] bf16,
// W [DIM][wstride] bf16. 256 threads = 8 waves, wave w computes rows m0+16w..+15.
// A tile staged global->LDS by TDM (USE_TDM) or async-to-LDS b128 copies; tile i+1
// copies are issued before computing tile i so WMMAs hide the DMA latency.
template <bool USE_TDM>
__device__ inline void gemm_mainloop(const bf16_t* __restrict__ A, const bf16_t* __restrict__ W,
                                     int wstride, int m0, int n0,
                                     bf16_t* As2 /*2x128x32*/, bf16_t* Bs2 /*2x64x32*/,
                                     AccF acc[4]) {
  const int tid = threadIdx.x;
  const int wave = tid >> 5, lane = tid & 31;
  // B loader: each thread covers 4 n-cols x 2 adjacent k-rows -> 4 packed b32 LDS stores
  const int nB = (tid & 15) * 4;
  const int kB = (tid >> 4) * 2;
  // async-path A mapping: 32B per thread
  const int rowA = tid >> 1, offcA = (tid & 1) * 16;
  const unsigned asBase = (unsigned)(uintptr_t)(void*)As2;
  const unsigned ldsA = (unsigned)(uintptr_t)(void*)(As2 + rowA * 32 + offcA);

  unsigned short bra[4], brb[4];

  auto issueA = [&](int k0, int buf) {
    if (USE_TDM) {
      if (tid < 32)  // TDM ignores EXEC: issue once, from wave 0 only
        tdm_load_tile_bf16(asBase + (unsigned)buf * (128 * 32 * 2),
                           A + (size_t)m0 * DIM + k0, 32, 128, DIM);
    } else {
      unsigned long long ga =
          (unsigned long long)(uintptr_t)(A + (size_t)(m0 + rowA) * DIM + k0 + offcA);
      unsigned ldst = ldsA + (unsigned)buf * (128 * 32 * 2);
      asm volatile("global_load_async_to_lds_b128 %0, %1, off"
                   :: "v"(ldst), "v"(ga) : "memory");
      asm volatile("global_load_async_to_lds_b128 %0, %1, off offset:16"
                   :: "v"(ldst), "v"(ga) : "memory");
    }
  };
  auto loadB = [&](int k0) {
    const unsigned short* p0 = (const unsigned short*)(W + (size_t)(k0 + kB) * wstride + n0 + nB);
    const unsigned short* p1 = (const unsigned short*)(W + (size_t)(k0 + kB + 1) * wstride + n0 + nB);
#pragma unroll
    for (int j = 0; j < 4; ++j) { bra[j] = p0[j]; brb[j] = p1[j]; }
  };
  auto storeB = [&](int buf) {  // Bs[n][k] transposed: pack {k,k+1} into one dword
    unsigned* dst = (unsigned*)(Bs2 + buf * (64 * 32));
#pragma unroll
    for (int j = 0; j < 4; ++j)
      dst[((nB + j) * 32 + kB) >> 1] = (unsigned)bra[j] | ((unsigned)brb[j] << 16);
  };
  auto waitCopies = [&]() {
    if (USE_TDM) {
      if (tid < 32) __builtin_amdgcn_s_wait_tensorcnt(0);
    } else {
      asm volatile("s_wait_asynccnt 0x0" ::: "memory");
    }
  };

  // prologue: tile 0 into buffer 0
  issueA(0, 0);
  loadB(0);
  storeB(0);
  waitCopies();
  __syncthreads();

  const int NK = DIM / 32;
  for (int it = 0; it < NK; ++it) {
    const int cur = it & 1;
    const bool more = (it + 1) < NK;
    if (more) {               // prefetch next tile into the other buffer
      issueA((it + 1) * 32, cur ^ 1);
      loadB((it + 1) * 32);
    }
    const bf16_t* Asb = As2 + cur * (128 * 32);
    const bf16_t* Bsb = Bs2 + cur * (64 * 32);
    v16bf af = frag_rowmajor(Asb + wave * 16 * 32, 32, lane);
#pragma unroll
    for (int nt = 0; nt < 4; ++nt) {
      v16bf bv = frag_rowmajor(Bsb + nt * 16 * 32, 32, lane);
      acc[nt].v = WMMA_BF16(af, bv, acc[nt].v);
    }
    if (more) storeB(cur ^ 1);
    waitCopies();
    __syncthreads();
  }
}

// QKV projection + fused RoPE (table-driven, +1/sqrt(HD) folded into Q).
// q,k scattered to [B,H,S,D] bf16; V stored TRANSPOSED as [B,H,D,S] bf16 so the
// attention PV matmul reads contiguous B-fragments. A tile staged via TDM.
__global__ __launch_bounds__(256) void k_gemm_qkv_rope(const bf16_t* __restrict__ A,
                                                       const bf16_t* __restrict__ W,
                                                       const float* __restrict__ cosT,
                                                       const float* __restrict__ sinT,
                                                       bf16_t* __restrict__ qout,
                                                       bf16_t* __restrict__ kout,
                                                       bf16_t* __restrict__ vtout) {
  __shared__ __align__(16) bf16_t As2[2 * 128 * 32];
  __shared__ __align__(16) bf16_t Bs2[2 * 64 * 32];
  const int m0 = blockIdx.x * 128;
  const int n0 = blockIdx.y * 64;  // 64-wide tile == exactly one head of one of q/k/v
  AccF acc[4];
#pragma unroll
  for (int i = 0; i < 4; ++i)
#pragma unroll
    for (int j = 0; j < 8; ++j) acc[i].f[j] = 0.f;

  gemm_mainloop<true>(A, W, NQKV, m0, n0, As2, Bs2, acc);

  const int wave = threadIdx.x >> 5, lane = threadIdx.x & 31;
  const int which = n0 / DIM;          // 0=q, 1=k, 2=v (uniform per block)
  const int h = (n0 % DIM) / HD;       // head (uniform per block)
  const int coll = lane & 15;
  const int rowoff = 8 * (lane >> 4);
#pragma unroll
  for (int r = 0; r < 8; ++r) {
    int t = m0 + wave * 16 + r + rowoff;
    int b = t >> 11;
    int s = t & (SEQ - 1);
#pragma unroll
    for (int nt = 0; nt < 4; ++nt) {
      int d = nt * 16 + coll;          // 0..63 within head
      float x = acc[nt].f[r];
      if (which == 2) {
        vtout[(((size_t)b * NH + h) * HD + d) * SEQ + s] = (bf16_t)x;  // V^T: [B,H,D,S]
      } else {
        int j = d & 31;
        float cs = cosT[s * 32 + j];
        float sn = sinT[s * 32 + j];
        float part = acc[nt ^ 2].f[r];   // rotate_half partner lives 2 tiles away
        float val = (d < 32) ? (x * cs - part * sn) : (x * cs + part * sn);
        if (which == 0) val *= 0.125f;   // 1/sqrt(64) folded into q
        bf16_t* dst = (which == 0) ? qout : kout;
        dst[(((size_t)b * NH + h) * SEQ + s) * HD + d] = (bf16_t)val;
      }
    }
  }
}

// Sliding-window flash attention: one wave per (b, h, 16-query block).
__global__ __launch_bounds__(32) void k_attn(const bf16_t* __restrict__ q,
                                             const bf16_t* __restrict__ k,
                                             const bf16_t* __restrict__ vt,
                                             bf16_t* __restrict__ ctx) {
  __shared__ __align__(16) bf16_t Plds[16 * 32];
  const int lane = threadIdx.x;
  const int bid = blockIdx.x;
  const int qb = bid & 127;            // SEQ/16 = 128
  const int h = (bid >> 7) % NH;
  const int b = bid / (128 * NH);
  const int q0 = qb * 16;
  const bf16_t* qh = q + ((size_t)b * NH + h) * SEQ * HD;
  const bf16_t* kh = k + ((size_t)b * NH + h) * SEQ * HD;
  const bf16_t* vh = vt + ((size_t)b * NH + h) * HD * SEQ;  // [D][S]
  const int coll = lane & 15;
  const int rowoff = 8 * (lane >> 4);

  v16bf qf0 = frag_rowmajor(qh + (size_t)q0 * HD, HD, lane);
  v16bf qf1 = frag_rowmajor(qh + (size_t)q0 * HD + 32, HD, lane);

  float mrun[8], lrun[8];
  AccF o[4];
#pragma unroll
  for (int r = 0; r < 8; ++r) { mrun[r] = -3.0e38f; lrun[r] = 0.f; }
#pragma unroll
  for (int i = 0; i < 4; ++i)
#pragma unroll
    for (int j = 0; j < 8; ++j) o[i].f[j] = 0.f;

  int kstart = q0 - WINDOW;
  if (kstart < 0) kstart = 0;
  kstart &= ~31;
  int kend = q0 + 15 + WINDOW + 1;
  kend = (kend + 31) & ~31;
  if (kend > SEQ) kend = SEQ;

  for (int kb0 = kstart; kb0 < kend; kb0 += 32) {
    // S = q @ k^T for 16 queries x 32 keys (two 16x16 C tiles, K=64 in two WMMA steps)
    AccF sa[2];
#pragma unroll
    for (int i = 0; i < 2; ++i)
#pragma unroll
      for (int j = 0; j < 8; ++j) sa[i].f[j] = 0.f;
#pragma unroll
    for (int nt = 0; nt < 2; ++nt) {
      const bf16_t* kb = kh + (size_t)(kb0 + nt * 16) * HD;  // B=K^T == [key][hd] row-major
      v16bf b0 = frag_rowmajor(kb, HD, lane);
      v16bf b1 = frag_rowmajor(kb + 32, HD, lane);
      sa[nt].v = WMMA_BF16(qf0, b0, sa[nt].v);
      sa[nt].v = WMMA_BF16(qf1, b1, sa[nt].v);
    }
    // mask + online softmax (row = 16 lanes of a half-wave at one VGPR slot)
    float alpha[8];
#pragma unroll
    for (int r = 0; r < 8; ++r) {
      int qi = q0 + r + rowoff;
#pragma unroll
      for (int nt = 0; nt < 2; ++nt) {
        int kj = kb0 + nt * 16 + coll;
        int dlt = qi - kj;
        if (dlt < 0) dlt = -dlt;
        if (dlt > WINDOW) sa[nt].f[r] += NEGBIG;
      }
      float t = fmaxf(sa[0].f[r], sa[1].f[r]);
#pragma unroll
      for (int m = 1; m < 16; m <<= 1) t = fmaxf(t, __shfl_xor(t, m, 32));
      float mnew = fmaxf(mrun[r], t);
      alpha[r] = __expf(mrun[r] - mnew);
      float ps = 0.f;
#pragma unroll
      for (int nt = 0; nt < 2; ++nt) {
        float p = __expf(sa[nt].f[r] - mnew);
        sa[nt].f[r] = p;
        ps += p;
      }
#pragma unroll
      for (int m = 1; m < 16; m <<= 1) ps += __shfl_xor(ps, m, 32);
      lrun[r] = lrun[r] * alpha[r] + ps;
      mrun[r] = mnew;
    }
#pragma unroll
    for (int i = 0; i < 4; ++i)
#pragma unroll
      for (int r = 0; r < 8; ++r) o[i].f[r] *= alpha[r];
    // Re-layout P (C-layout f32) -> A-fragment via LDS, bf16 [16 rows][32 keys]
#pragma unroll
    for (int nt = 0; nt < 2; ++nt)
#pragma unroll
      for (int r = 0; r < 8; ++r)
        Plds[(r + rowoff) * 32 + nt * 16 + coll] = (bf16_t)sa[nt].f[r];
    asm volatile("s_wait_dscnt 0x0" ::: "memory");  // cross-lane LDS reuse within wave
    v16bf pf = frag_rowmajor(Plds, 32, lane);
#pragma unroll
    for (int nt = 0; nt < 4; ++nt) {
      // V^T is [D][S]: N-major for the B operand -> contiguous b32-pair fragment loads
      v16bf vf = frag_rowmajor(vh + (size_t)(nt * 16) * SEQ + kb0, SEQ, lane);
      o[nt].v = WMMA_BF16(pf, vf, o[nt].v);
    }
    asm volatile("s_wait_dscnt 0x0" ::: "memory");  // drain before next tile overwrites Plds
  }
  // normalize and write context as bf16 [B][S][DIM]
#pragma unroll
  for (int nt = 0; nt < 4; ++nt)
#pragma unroll
    for (int r = 0; r < 8; ++r) {
      int s = q0 + r + rowoff;
      float val = o[nt].f[r] / lrun[r];
      ctx[((size_t)b * SEQ + s) * DIM + h * HD + nt * 16 + coll] = (bf16_t)val;
    }
}

// Output projection: out_f32[4096][768] = ctx_bf16 @ Wo_bf16 (async-to-LDS A path)
__global__ __launch_bounds__(256) void k_gemm_out(const bf16_t* __restrict__ A,
                                                  const bf16_t* __restrict__ W,
                                                  float* __restrict__ out) {
  __shared__ __align__(16) bf16_t As2[2 * 128 * 32];
  __shared__ __align__(16) bf16_t Bs2[2 * 64 * 32];
  const int m0 = blockIdx.x * 128;
  const int n0 = blockIdx.y * 64;
  AccF acc[4];
#pragma unroll
  for (int i = 0; i < 4; ++i)
#pragma unroll
    for (int j = 0; j < 8; ++j) acc[i].f[j] = 0.f;

  gemm_mainloop<false>(A, W, DIM, m0, n0, As2, Bs2, acc);

  const int wave = threadIdx.x >> 5, lane = threadIdx.x & 31;
  const int coll = lane & 15;
  const int rowoff = 8 * (lane >> 4);
#pragma unroll
  for (int r = 0; r < 8; ++r) {
    int t = m0 + wave * 16 + r + rowoff;
#pragma unroll
    for (int nt = 0; nt < 4; ++nt)
      out[(size_t)t * DIM + n0 + nt * 16 + coll] = acc[nt].f[r];
  }
}

extern "C" void kernel_launch(void* const* d_in, const int* in_sizes, int n_in,
                              void* d_out, int out_size, void* d_ws, size_t ws_size,
                              hipStream_t stream) {
  const float* hidden = (const float*)d_in[0];
  const float* Wqkv = (const float*)d_in[1];
  const float* Wo = (const float*)d_in[2];
  float* out = (float*)d_out;

  char* ws = (char*)d_ws;
  size_t off = 0;
  auto carve = [&](size_t bytes) -> void* {
    void* p = ws + off;
    off += (bytes + 255) & ~(size_t)255;
    return p;
  };
  bf16_t* hbf    = (bf16_t*)carve((size_t)BS * SEQ * DIM * sizeof(bf16_t));
  bf16_t* wqkvbf = (bf16_t*)carve((size_t)DIM * NQKV * sizeof(bf16_t));
  bf16_t* wobf   = (bf16_t*)carve((size_t)DIM * DIM * sizeof(bf16_t));
  bf16_t* qb     = (bf16_t*)carve((size_t)BS * NH * SEQ * HD * sizeof(bf16_t));
  bf16_t* kb     = (bf16_t*)carve((size_t)BS * NH * SEQ * HD * sizeof(bf16_t));
  bf16_t* vtb    = (bf16_t*)carve((size_t)BS * NH * HD * SEQ * sizeof(bf16_t));
  bf16_t* ctx    = (bf16_t*)carve((size_t)BS * SEQ * DIM * sizeof(bf16_t));
  float*  cosT   = (float*)carve((size_t)SEQ * 32 * sizeof(float));
  float*  sinT   = (float*)carve((size_t)SEQ * 32 * sizeof(float));
  (void)ws_size; (void)in_sizes; (void)n_in; (void)out_size;

  k_rope_tables<<<(SEQ * 32) / 256, 256, 0, stream>>>(cosT, sinT);
  int n1 = BS * SEQ * DIM;
  k_cvt_f32_bf16<<<(n1 + 255) / 256, 256, 0, stream>>>(hidden, hbf, n1);
  int n2 = DIM * NQKV;
  k_cvt_f32_bf16<<<(n2 + 255) / 256, 256, 0, stream>>>(Wqkv, wqkvbf, n2);
  int n3 = DIM * DIM;
  k_cvt_f32_bf16<<<(n3 + 255) / 256, 256, 0, stream>>>(Wo, wobf, n3);

  k_gemm_qkv_rope<<<dim3(BS * SEQ / 128, NQKV / 64), 256, 0, stream>>>(hbf, wqkvbf, cosT, sinT,
                                                                       qb, kb, vtb);
  k_attn<<<BS * NH * (SEQ / 16), 32, 0, stream>>>(qb, kb, vtb, ctx);
  k_gemm_out<<<dim3(BS * SEQ / 128, DIM / 64), 256, 0, stream>>>(ctx, wobf, out);
}
